// DetectionGenerator_34291018891349
// MI455X (gfx1250) — compile-verified, matrix operand order checked by reading, not run on hardware
//
#include <hip/hip_runtime.h>
#include <hip/hip_bf16.h>
#include <stdint.h>

// Problem constants (match reference)
#define Bn    8
#define Nn    8192
#define Cn    91
#define CMn   90
#define Kn    1000
#define MAXT  100
#define IOU_THRESH 0.3f
#define BBOX_CLIPV 4.135166556742356f   // log(1000/16)

typedef __attribute__((ext_vector_type(16))) _Float16 v16h;
typedef __attribute__((ext_vector_type(8)))  float    v8f;
typedef __attribute__((ext_vector_type(4)))  unsigned v4u;
typedef __attribute__((ext_vector_type(8)))  int      v8i;
typedef __attribute__((ext_vector_type(4)))  int      v4i;

// ---- CDNA5 async global->LDS copies (ASYNCcnt tracked) -------------------
__device__ __forceinline__ void async_load_b32(unsigned lds_off, const void* g) {
    asm volatile("global_load_async_to_lds_b32 %0, %1, off"
                 :: "v"(lds_off), "v"(g) : "memory");
}
__device__ __forceinline__ void async_load_b128(unsigned lds_off, const void* g) {
    asm volatile("global_load_async_to_lds_b128 %0, %1, off"
                 :: "v"(lds_off), "v"(g) : "memory");
}
__device__ __forceinline__ void wait_async0() {
    asm volatile("s_wait_asynccnt 0" ::: "memory");
}
__device__ __forceinline__ unsigned lds_off_of(const void* p) {
    // LDS flat addresses carry the byte offset in the low 32 bits (CDNA5 aperture rule)
    return (unsigned)(uintptr_t)p;
}

// ==========================================================================
// Kernel 1: softmax scores -> scores_t[b][c-1][n]  (transposed, per-class rows)
// One wave handles a 16-anchor x 96-class (padded) tile.  Row sums of exp()
// run on the matrix pipe: 3x v_wmma_f32_16x16x32_f16 against a ones matrix.
// Pad columns hold -1e30 so exp() underflows to the matmul-neutral 0 -- no
// per-lane guards, fully straight-line fragment build.
// ==========================================================================
__global__ __launch_bounds__(256) void k_softmax(const float* __restrict__ cls,
                                                 float* __restrict__ scores_t) {
    __shared__ __align__(16) float s_tile[8][16][96];
    __shared__ float s_rmax[8][16];
    __shared__ float s_rinv[8][16];

    const int wv  = threadIdx.x >> 5;
    const int l   = threadIdx.x & 31;
    const int hf  = l >> 4;          // half-wave select (K-fragment split)
    const int m   = l & 15;          // matrix row handled by this lane
    const long long g0 = ((long long)blockIdx.x * 8 + wv) * 16;   // global anchor base
    const int b    = (int)(g0 / Nn);
    const int nloc = (int)(g0 % Nn);

    // ---- stream 16x91 logits into LDS via async-to-LDS (per-lane scatter) ----
    const unsigned lbase = lds_off_of(&s_tile[wv][0][0]);
    for (int step = 0; step < 46; ++step) {
        int t = step * 32 + l;
        if (t < 16 * 91) {
            int row = t / 91, col = t - row * 91;
            async_load_b32(lbase + (unsigned)(row * 96 + col) * 4u,
                           cls + (size_t)(g0 + row) * Cn + col);
        }
    }
    wait_async0();
    // pad columns 91..95: exp(pad - rmax) == 0 under f32->f16, neutral for the sum
    for (int t = l; t < 16 * 5; t += 32) {
        int row = t / 5, col = 91 + t % 5;
        s_tile[wv][row][col] = -1e30f;
    }
    __syncthreads();

    // ---- per-row max (stable softmax), vectorized LDS reads ----
    const float* rp = &s_tile[wv][m][0];
    float rmax = -1e30f;
    for (int c4 = 0; c4 < 96; c4 += 4) {
        float4 v = *reinterpret_cast<const float4*>(rp + c4);
        rmax = fmaxf(rmax, fmaxf(fmaxf(v.x, v.y), fmaxf(v.z, v.w)));
    }

    // ---- row sums of exp() via WMMA:  D = A(16x32 f16) * Ones(32x16) + C ----
    v16h ones;
#pragma unroll
    for (int e = 0; e < 16; ++e) ones[e] = (_Float16)1.0f;
    v8f acc = {};
#pragma unroll
    for (int kc = 0; kc < 3; ++kc) {
        v16h a;
#pragma unroll
        for (int e = 0; e < 16; ++e) {
            // 16-bit A layout: lanes 0-15 hold K=0..7,16..23; lanes 16-31 K=8..15,24..31
            int kk = kc * 32 + ((e < 8) ? (hf * 8 + e) : (16 + hf * 8 + (e - 8)));
            a[e] = (_Float16)__expf(rp[kk] - rmax);   // kk <= 95, pad -> 0
        }
        acc = __builtin_amdgcn_wmma_f32_16x16x32_f16(false, a, false, ones,
                                                     (short)0, acc, false, false);
    }
    // D layout: lane with col==0 holds rows r+8*hf in regs 0..7 -> broadcast via LDS
    if (m == 0) {
#pragma unroll
        for (int r = 0; r < 8; ++r) s_rinv[wv][r + 8 * hf] = 1.0f / acc[r];
    }
    if (l < 16) s_rmax[wv][l] = rmax;
    __syncthreads();

    // ---- write normalized per-class scores, transposed: [b][c-1][n] ----
    const float rm = s_rmax[wv][m];
    const float ri = s_rinv[wv][m];
    for (int cc = 0; cc < 45; ++cc) {
        int c = 1 + cc * 2 + hf;                       // classes 1..90
        float v = __expf(s_tile[wv][m][c] - rm) * ri;
        scores_t[((size_t)b * CMn + (c - 1)) * Nn + nloc + m] = v;
    }
}

// ==========================================================================
// Kernel 2: per-(b,c) top-1000 of 8192 scores.  The contiguous 32KB score row
// is DMA'd into LDS with ONE Tensor Data Mover descriptor (TENSORcnt), then
// bitonic-sorted in LDS (keys flipped so ascending == score descending).
// ==========================================================================
__global__ __launch_bounds__(256) void k_topk(const float* __restrict__ scores_t,
                                              int* __restrict__ topk_idx,
                                              float* __restrict__ topk_score) {
    __shared__ __align__(16) unsigned s_key[8192];
    __shared__ unsigned short s_idx[8192];
    const int tid = threadIdx.x;
    const int bc  = blockIdx.x;
    const float* src = scores_t + (size_t)bc * Nn;

    if (tid < 32) {   // wave 0 issues the TDM descriptor (EXEC ignored by TDM)
        uint64_t ga = (uint64_t)(uintptr_t)src;
        v4u g0;
        g0[0] = 1u;                                    // count=1: valid descriptor
        g0[1] = lds_off_of(&s_key[0]);                 // lds_addr
        g0[2] = (unsigned)(ga & 0xFFFFFFFFu);          // global_addr[31:0]
        g0[3] = (unsigned)((ga >> 32) & 0x1FFFFFFu)    // global_addr[56:32]
                | (2u << 30);                          // type=2 ("image")
        v8i g1;
        g1[0] = 0x00020000;   // data_size=2 (4B), no multicast, no pad/iterate
        g1[1] = 0x20000000;   // tensor_dim0 = 8192 (bits 79:48, low half here)
        g1[2] = 0x00010000;   // tensor_dim0 hi = 0, tensor_dim1 = 1
        g1[3] = 0x20000000;   // tile_dim0 = 8192 (bits 127:112)
        g1[4] = 0x00000001;   // tile_dim1 = 1
        g1[5] = 0x00002000;   // tensor_dim0_stride = 8192
        g1[6] = 0;
        g1[7] = 0;
        v4i z4 = {0, 0, 0, 0};
        v8i z8 = {0, 0, 0, 0, 0, 0, 0, 0};
        __builtin_amdgcn_tensor_load_to_lds(g0, g1, z4, z4, z8, 0);
        __builtin_amdgcn_s_wait_tensorcnt(0);
    }
    __syncthreads();

    for (int t = 0; t < 32; ++t) {
        int i = tid + t * 256;
        unsigned u = s_key[i];                              // raw float bits
        unsigned mapped = (u & 0x80000000u) ? ~u : (u | 0x80000000u);
        s_key[i] = ~mapped;                                 // ascending -> score desc
        s_idx[i] = (unsigned short)i;
    }
    __syncthreads();

    for (int k = 2; k <= 8192; k <<= 1) {
        for (int j = k >> 1; j > 0; j >>= 1) {
            for (int t = 0; t < 32; ++t) {
                int i = tid + t * 256;
                int p = i ^ j;
                if (p > i) {
                    bool up = ((i & k) == 0);
                    unsigned ki = s_key[i], kp = s_key[p];
                    if ((ki > kp) == up) {
                        s_key[i] = kp; s_key[p] = ki;
                        unsigned short tt = s_idx[i]; s_idx[i] = s_idx[p]; s_idx[p] = tt;
                    }
                }
            }
            __syncthreads();
        }
    }
    for (int t = 0; t < 4; ++t) {
        int k = tid + t * 256;
        if (k < Kn) {
            int idx = s_idx[k];
            topk_idx[(size_t)bc * Kn + k]   = idx;
            topk_score[(size_t)bc * Kn + k] = src[idx];
        }
    }
}

// ==========================================================================
// Kernel 3: per-(b,c) box decode (only the 1000 survivors!) + greedy NMS.
// One wave per block; boxes live in LDS; wave32 in-order DS ops provide the
// barrier-free shared suppression flag.
// ==========================================================================
__global__ __launch_bounds__(32) void k_nms(const float* __restrict__ box_out,
                                            const float* __restrict__ anchors,
                                            const float* __restrict__ image_info,
                                            const int* __restrict__ topk_idx,
                                            const float* __restrict__ topk_score,
                                            float* __restrict__ nmsbox,
                                            float* __restrict__ nmsscore) {
    __shared__ float s_box[Kn][4];
    __shared__ float s_sc[Kn];
    __shared__ volatile unsigned s_keep[32];
    __shared__ volatile int s_flag;

    const int bc = blockIdx.x;
    const int b  = bc / CMn;
    const int jc = bc - b * CMn;         // class index in scores (0..89)
    const int cfull = jc + 1;
    const int l = threadIdx.x;
    const float H = image_info[b * 4 + 0];
    const float W = image_info[b * 4 + 1];
    const float invH = 1.0f / H, invW = 1.0f / W;

    for (int k = l; k < Kn; k += 32) {
        int n = topk_idx[(size_t)bc * Kn + k];
        const float* e = box_out + (size_t)(b * Nn + n) * (Cn * 4) + cfull * 4;
        const float* a = anchors + (size_t)(b * Nn + n) * 4;
        __builtin_prefetch(e, 0, 1);
        float e0 = e[0], e1 = e[1], e2 = e[2], e3 = e[3];
        float a0 = a[0], a1 = a[1], a2 = a[2], a3 = a[3];
        float ah = a2 - a0, aw = a3 - a1;
        float acy = a0 + 0.5f * ah, acx = a1 + 0.5f * aw;
        float ty = e0 * 0.1f, tx = e1 * 0.1f;
        float th = fminf(e2 * 0.2f, BBOX_CLIPV), tw = fminf(e3 * 0.2f, BBOX_CLIPV);
        float cy = ty * ah + acy, cx = tx * aw + acx;
        float h = __expf(th) * ah, w = __expf(tw) * aw;
        float y0 = cy - 0.5f * h, x0 = cx - 0.5f * w;
        float y1 = cy + 0.5f * h, x1 = cx + 0.5f * w;
        y0 = fminf(fmaxf(y0, 0.f), H) * invH;  y1 = fminf(fmaxf(y1, 0.f), H) * invH;
        x0 = fminf(fmaxf(x0, 0.f), W) * invW;  x1 = fminf(fmaxf(x1, 0.f), W) * invW;
        s_box[k][0] = y0; s_box[k][1] = x0; s_box[k][2] = y1; s_box[k][3] = x1;
        s_sc[k] = topk_score[(size_t)bc * Kn + k];
        float* ob = nmsbox + ((size_t)bc * Kn + k) * 4;
        ob[0] = y0; ob[1] = x0; ob[2] = y1; ob[3] = x1;
    }
    s_keep[l] = 0u;    // same wave: DS ops stay in order, no barrier needed

    for (int i = 0; i < Kn; ++i) {
        float iy0 = s_box[i][0], ix0 = s_box[i][1], iy1 = s_box[i][2], ix1 = s_box[i][3];
        float iarea = fmaxf(iy1 - iy0, 0.f) * fmaxf(ix1 - ix0, 0.f);
        bool loc = false;
        for (int q = l; q < i; q += 32) {
            if ((s_keep[q >> 5] >> (q & 31)) & 1u) {
                float qy0 = s_box[q][0], qx0 = s_box[q][1];
                float qy1 = s_box[q][2], qx1 = s_box[q][3];
                float qarea = fmaxf(qy1 - qy0, 0.f) * fmaxf(qx1 - qx0, 0.f);
                float ih = fmaxf(fminf(qy1, iy1) - fmaxf(qy0, iy0), 0.f);
                float iw = fmaxf(fminf(qx1, ix1) - fmaxf(qx0, ix0), 0.f);
                float inter = ih * iw;
                if (inter / (qarea + iarea - inter + 1e-8f) > IOU_THRESH) loc = true;
            }
        }
        if (l == 0) s_flag = 0;        // wave-wide any() through LDS (in-order DS)
        if (loc)    s_flag = 1;
        int sup = s_flag;
        if (l == 0 && s_sc[i] > 0.0f && !sup) s_keep[i >> 5] |= (1u << (i & 31));
    }

    for (int k = l; k < Kn; k += 32) {
        bool kp = (s_keep[k >> 5] >> (k & 31)) & 1u;
        nmsscore[(size_t)bc * Kn + k] = kp ? s_sc[k] : -1.0f;
    }
}

// ==========================================================================
// Kernel 4: per-image top-100 of 90*1000 post-NMS scores.  Scores are staged
// through LDS with async-b128 chunks, positives compacted (atomic counter),
// then a 4096-key bitonic sort emits the final detections.
// ==========================================================================
__global__ __launch_bounds__(256) void k_final(const float* __restrict__ image_info,
                                               const float* __restrict__ nmsbox,
                                               const float* __restrict__ nmsscore,
                                               float* __restrict__ out) {
    __shared__ unsigned s_key[4096];
    __shared__ unsigned s_pay[4096];
    __shared__ __align__(16) float s_stage[4096];
    __shared__ int s_cnt;
    const int b = blockIdx.x;
    const int tid = threadIdx.x;
    if (tid == 0) s_cnt = 0;
    __syncthreads();

    const float* sc = nmsscore + (size_t)b * CMn * Kn;
    const unsigned sbase = lds_off_of(&s_stage[0]);
    for (int ch = 0; ch < 22; ++ch) {                // 22*4096 >= 90000
        int base = ch * 4096;
        for (int q = 0; q < 4; ++q) {
            int e = base + tid * 4 + q * 1024;       // 16B per lane (90000 % 4 == 0)
            if (e < CMn * Kn)
                async_load_b128(sbase + (unsigned)(e - base) * 4u, sc + e);
        }
        wait_async0();
        __syncthreads();
        int lim = CMn * Kn - base; if (lim > 4096) lim = 4096;
        for (int i = tid; i < lim; i += 256) {
            float f = s_stage[i];
            if (f > 0.0f) {
                int slot = atomicAdd(&s_cnt, 1);
                if (slot < 4096) {
                    unsigned u = __float_as_uint(f);
                    unsigned mapped = (u & 0x80000000u) ? ~u : (u | 0x80000000u);
                    s_key[slot] = ~mapped;
                    s_pay[slot] = (unsigned)(base + i);      // i = c*1000 + k
                }
            }
        }
        __syncthreads();
    }
    int cnt = s_cnt; if (cnt > 4096) cnt = 4096;
    for (int i = tid; i < 4096; i += 256)
        if (i >= cnt) { s_key[i] = 0xFFFFFFFFu; s_pay[i] = 0u; }
    __syncthreads();

    for (int k = 2; k <= 4096; k <<= 1) {
        for (int j = k >> 1; j > 0; j >>= 1) {
            for (int t = 0; t < 16; ++t) {
                int i = tid + t * 256;
                int p = i ^ j;
                if (p > i) {
                    bool up = ((i & k) == 0);
                    unsigned ki = s_key[i], kp = s_key[p];
                    if ((ki > kp) == up) {
                        s_key[i] = kp; s_key[p] = ki;
                        unsigned tp = s_pay[i]; s_pay[i] = s_pay[p]; s_pay[p] = tp;
                    }
                }
            }
            __syncthreads();
        }
    }

    const float H = image_info[b * 4 + 0], W = image_info[b * 4 + 1];
    float* out_nv  = out;                                 // [8]
    float* out_box = out + Bn;                            // [8][100][4]
    float* out_cls = out + Bn + Bn * MAXT * 4;            // [8][100]
    float* out_sc  = out + Bn + Bn * MAXT * 4 + Bn * MAXT;// [8][100]
    if (tid < MAXT) {
        bool valid = tid < cnt;
        unsigned pay = s_pay[tid];
        float scv = valid ? sc[pay] : 0.0f;
        int   cls = valid ? (int)(pay / Kn) : 0;
        out_sc[b * MAXT + tid]  = scv;
        out_cls[b * MAXT + tid] = (float)(cls + 1);
        const float* nb = nmsbox + ((size_t)b * CMn * Kn + pay) * 4;
        float* ob = out_box + ((size_t)b * MAXT + tid) * 4;
        ob[0] = valid ? nb[0] * H : 0.0f;
        ob[1] = valid ? nb[1] * W : 0.0f;
        ob[2] = valid ? nb[2] * H : 0.0f;
        ob[3] = valid ? nb[3] * W : 0.0f;
    }
    if (tid == 0) out_nv[b] = (float)(cnt < MAXT ? cnt : MAXT);
}

// ==========================================================================
extern "C" void kernel_launch(void* const* d_in, const int* in_sizes, int n_in,
                              void* d_out, int out_size, void* d_ws, size_t ws_size,
                              hipStream_t stream) {
    (void)in_sizes; (void)n_in; (void)out_size; (void)ws_size;
    const float* cls  = (const float*)d_in[0];   // class_outputs [8,8192,91]
    const float* boxo = (const float*)d_in[1];   // box_outputs   [8,8192,364]
    const float* anch = (const float*)d_in[2];   // anchor_boxes  [8,8192,4]
    const float* info = (const float*)d_in[3];   // image_info    [8,4]

    char* w = (char*)d_ws;                               // ~43.7 MB total
    float* scores_t   = (float*)w;  w += (size_t)Bn * CMn * Nn * 4;   // 23.6 MB
    int*   topk_idx   = (int*)w;    w += (size_t)Bn * CMn * Kn * 4;   // 2.88 MB
    float* topk_score = (float*)w;  w += (size_t)Bn * CMn * Kn * 4;   // 2.88 MB
    float* nmsbox     = (float*)w;  w += (size_t)Bn * CMn * Kn * 16;  // 11.5 MB
    float* nmsscore   = (float*)w;  w += (size_t)Bn * CMn * Kn * 4;   // 2.88 MB

    k_softmax<<<(Bn * Nn) / (16 * 8), 256, 0, stream>>>(cls, scores_t);
    k_topk   <<<Bn * CMn,            256, 0, stream>>>(scores_t, topk_idx, topk_score);
    k_nms    <<<Bn * CMn,             32, 0, stream>>>(boxo, anch, info, topk_idx,
                                                       topk_score, nmsbox, nmsscore);
    k_final  <<<Bn,                  256, 0, stream>>>(info, nmsbox, nmsscore,
                                                       (float*)d_out);
}
